// TripletLoss_2688649527615
// MI455X (gfx1250) — matrix operand order repaired and checked by reading, not compiled, not run on hardware
//
#include <hip/hip_runtime.h>

// ---------------------------------------------------------------------------
// Cosine-similarity hardest-pos/hardest-neg triplet loss, MI455X (gfx1250).
//
// sim = fn @ fn^T: 34.4 GFLOP over 4 MB of f16 data -> compute/L2-bound;
// the 256 MB sim matrix is never materialized (min/max fused per 16x16 tile).
// Reuse: wave computes M=32 (two A tiles in VGPRs, two independent v_wmma
// chains per B fragment); 8 waves/block share one 16-col x 256-K B panel in
// LDS (double-buffered, register prefetch); columns split into 8 slices for
// occupancy (2048 waves). Total L2 traffic ~128 MB.
//
// Scheduling state (verified in round-3 disasm): ds loads pipelined ahead of
// WMMAs (partial dscnt waits), panel prefetch loadcnt wait at loop tail.
// Round-4 fix: fminf/fmaxf forced LLVM to emit v_max_num(x,x) NaN-
// canonicalization on all 32 running min/max registers every iteration;
// replace with direct v_min_num/v_max_num via inline asm (values are finite
// by construction), trimming the serialized VALU tail after the last WMMA.
// ---------------------------------------------------------------------------

typedef __attribute__((ext_vector_type(16))) _Float16 v16h;
typedef __attribute__((ext_vector_type(8)))  _Float16 v8h;
typedef __attribute__((ext_vector_type(8)))  float    v8f;
typedef __attribute__((ext_vector_type(4)))  int      v4i;

#define NROWS   8192
#define DIM     256
#define MARGIN  0.25f
#define FBIG    3.402823466e38f

#define ROWS_PER_BLOCK 256                       // 8 waves x 32 rows
#define CSLICES        8                         // column slices (blockIdx.y)
#define TILES_PER_SLICE (NROWS / 16 / CSLICES)   // 64 column tiles per slice
#define LDSROW 528                               // 512 B row + 16 B pad (bank-conflict-free)

union AV { v16h v; v8h h[2]; };

static __device__ __forceinline__ v8f wmma_f16(const v16h& a, const v16h& b, const v8f& c) {
  return __builtin_amdgcn_wmma_f32_16x16x32_f16(false, a, false, b, (short)0, c, false, false);
}

// Direct min/max without LLVM's sNaN canonicalization (inputs are finite).
static __device__ __forceinline__ float fmin_raw(float a, float b) {
  float d; asm("v_min_num_f32 %0, %1, %2" : "=v"(d) : "v"(a), "v"(b)); return d;
}
static __device__ __forceinline__ float fmax_raw(float a, float b) {
  float d; asm("v_max_num_f32 %0, %1, %2" : "=v"(d) : "v"(a), "v"(b)); return d;
}

// ---------------------------------------------------------------------------
// Kernel 1: row L2-normalize fp32 -> f16.  One wave32 per row.
// ---------------------------------------------------------------------------
__global__ __launch_bounds__(256) void normalize_f16(
    const float* __restrict__ f, _Float16* __restrict__ fh) {
  const int row  = (blockIdx.x * blockDim.x + threadIdx.x) >> 5;
  const int lane = threadIdx.x & 31;
  const float* rp = f + (size_t)row * DIM + lane * 8;

  float vals[8];
  float ss = 0.f;
#pragma unroll
  for (int i = 0; i < 8; ++i) { vals[i] = rp[i]; ss += vals[i] * vals[i]; }
#pragma unroll
  for (int m = 16; m >= 1; m >>= 1) ss += __shfl_xor(ss, m, 32);
  const float inv = rsqrtf(ss);

  _Float16* op = fh + (size_t)row * DIM + lane * 8;
#pragma unroll
  for (int i = 0; i < 8; ++i) op[i] = (_Float16)(vals[i] * inv);
}

// ---------------------------------------------------------------------------
// Kernel 2: main fused GEMM + running hardest-pos/neg.
// grid = (32 rowGroups, 8 colSlices), 256 threads (8 waves).
// Wave w owns rows rowGroup*256 + w*32 .. +31.
// ---------------------------------------------------------------------------
__global__ __launch_bounds__(256) void triplet_tile(
    const _Float16* __restrict__ fh, const int* __restrict__ labels,
    float* __restrict__ posPart, float* __restrict__ negPart) {
  __shared__ __align__(16) char smem[2][16 * LDSROW];   // double-buffered B panel
  __shared__ int sLab[CSLICES * TILES_PER_SLICE * 2];   // 1024 column labels of slice

  const int tid      = threadIdx.x;
  const int wave     = tid >> 5;
  const int lane     = tid & 31;
  const int laneHalf = lane >> 4;
  const int lane16   = lane & 15;
  const int r0       = blockIdx.x * ROWS_PER_BLOCK + wave * 32;
  const int slice    = blockIdx.y;
  const int ct0      = slice * TILES_PER_SLICE;

  // ---- A fragments: two 16x256 tiles, 8 K-chunks of 32 each, in VGPRs -----
  // ISA 16-bit A 16x32 layout: lanes 0-15 hold K=[0..7]+[16..23];
  // lanes 16-31 hold K=[8..15]+[24..31] for the same M = lane%16.
  v16h a0[8], a1[8];
  {
    const _Float16* ar0 = fh + (size_t)(r0 + lane16) * DIM;
    const _Float16* ar1 = fh + (size_t)(r0 + 16 + lane16) * DIM;
#pragma unroll
    for (int kk = 0; kk < 8; ++kk) {
      const int k0 = kk * 32 + laneHalf * 8;
      AV t;
      t.h[0] = *(const v8h*)(ar0 + k0);
      t.h[1] = *(const v8h*)(ar0 + k0 + 16);
      a0[kk] = t.v;
      t.h[0] = *(const v8h*)(ar1 + k0);
      t.h[1] = *(const v8h*)(ar1 + k0 + 16);
      a1[kk] = t.v;
    }
  }

  // Row labels for each accumulator slot (C layout: M = v + 8*laneHalf).
  int rowLab0[8], rowLab1[8];
#pragma unroll
  for (int v = 0; v < 8; ++v) {
    rowLab0[v] = labels[r0 + v + laneHalf * 8];
    rowLab1[v] = labels[r0 + 16 + v + laneHalf * 8];
  }

  float posMin[2][8], negMax[2][8];
#pragma unroll
  for (int v = 0; v < 8; ++v) {
    posMin[0][v] = FBIG;  posMin[1][v] = FBIG;
    negMax[0][v] = -FBIG; negMax[1][v] = -FBIG;
  }

  // ---- startup staging: slice's 1024 column labels + first B panel --------
  const int crow = tid >> 5;              // 0..7
  const int coff = (tid & 31) * 16;       // byte offset within row
  {
    *(v4i*)(sLab + tid * 4) = *(const v4i*)(labels + ct0 * 16 + tid * 4);

    const char* src = (const char*)(fh + (size_t)(ct0 * 16) * DIM);
    v8h c0 = *(const v8h*)(src + tid * 16);
    v8h c1 = *(const v8h*)(src + 4096 + tid * 16);
    *(v8h*)(smem[0] + crow * LDSROW + coff)       = c0;
    *(v8h*)(smem[0] + (8 + crow) * LDSROW + coff) = c1;
  }
  __syncthreads();

  int cur = 0;
  for (int it = 0; it < TILES_PER_SLICE; ++it) {
    // -- prefetch next B panel into registers (overlaps with the WMMAs) -----
    const int itn = (it + 1 < TILES_PER_SLICE) ? it + 1 : it;
    const char* nsrc = (const char*)(fh + (size_t)((ct0 + itn) * 16) * DIM);
    v8h p0 = *(const v8h*)(nsrc + tid * 16);
    v8h p1 = *(const v8h*)(nsrc + 4096 + tid * 16);

    const int colLab = sLab[it * 16 + lane16];          // ds_load_b32

    // -- 16 WMMAs over K=256, B fragments pipelined one K-step ahead --------
    // B 32x16 lane layout: lane -> column N = lane%16; lanes 0-15 carry
    // K=[0..7]+[16..23], lanes 16-31 carry K=[8..15]+[24..31] (matches A).
    const char* bbase = smem[cur] + lane16 * LDSROW + laneHalf * 16;
    AV b[2];
    b[0].h[0] = *(const v8h*)(bbase);
    b[0].h[1] = *(const v8h*)(bbase + 32);
    v8f acc0 = {}, acc1 = {};
#pragma unroll
    for (int kk = 0; kk < 8; ++kk) {
      if (kk < 7) {
        const int nxt = (kk + 1) & 1;
        b[nxt].h[0] = *(const v8h*)(bbase + (kk + 1) * 64);
        b[nxt].h[1] = *(const v8h*)(bbase + (kk + 1) * 64 + 32);
      }
      acc0 = wmma_f16(a0[kk], b[kk & 1].v, acc0);
      acc1 = wmma_f16(a1[kk], b[kk & 1].v, acc1);
    }

    // -- epilogue: classify the 16 sims this lane owns ----------------------
    // cndmask to +/-FBIG, then raw v_min/v_max (no canonicalization ops).
#pragma unroll
    for (int v = 0; v < 8; ++v) {
      const bool same0 = (colLab == rowLab0[v]);
      const bool same1 = (colLab == rowLab1[v]);
      posMin[0][v] = fmin_raw(posMin[0][v], same0 ? acc0[v] : FBIG);
      negMax[0][v] = fmax_raw(negMax[0][v], same0 ? -FBIG : acc0[v]);
      posMin[1][v] = fmin_raw(posMin[1][v], same1 ? acc1[v] : FBIG);
      negMax[1][v] = fmax_raw(negMax[1][v], same1 ? -FBIG : acc1[v]);
    }

    // -- commit prefetched panel to the other buffer ------------------------
    // (buffer cur^1 was last read in iteration it-1; the barrier that closed
    // it-1 makes this store safe, and this barrier publishes it.)
    *(v8h*)(smem[cur ^ 1] + crow * LDSROW + coff)       = p0;
    *(v8h*)(smem[cur ^ 1] + (8 + crow) * LDSROW + coff) = p1;
    __syncthreads();
    cur ^= 1;
  }

  // ---- reduce across the 16 lanes of each half (N dimension) --------------
#pragma unroll
  for (int s = 0; s < 2; ++s) {
#pragma unroll
    for (int v = 0; v < 8; ++v) {
#pragma unroll
      for (int m = 8; m >= 1; m >>= 1) {
        posMin[s][v] = fmin_raw(posMin[s][v], __shfl_xor(posMin[s][v], m, 32));
        negMax[s][v] = fmax_raw(negMax[s][v], __shfl_xor(negMax[s][v], m, 32));
      }
    }
  }

  // ---- write per-(slice,row) partials -------------------------------------
  if (lane16 == 0) {
    const size_t base = (size_t)slice * NROWS + r0;
#pragma unroll
    for (int v = 0; v < 8; ++v) {
      const int m = v + laneHalf * 8;
      posPart[base + m]      = posMin[0][v];
      negPart[base + m]      = negMax[0][v];
      posPart[base + 16 + m] = posMin[1][v];
      negPart[base + 16 + m] = negMax[1][v];
    }
  }
}

// ---------------------------------------------------------------------------
// Kernel 3: combine the 8 column slices per row, compute relu(neg-pos+margin),
// block-reduce. grid = 32 blocks x 256 threads (one thread per row).
// ---------------------------------------------------------------------------
__global__ __launch_bounds__(256) void row_loss(
    const float* __restrict__ posPart, const float* __restrict__ negPart,
    float* __restrict__ partials) {
  const int tid = threadIdx.x;
  const int row = blockIdx.x * 256 + tid;

  float p = FBIG, n = -FBIG;
#pragma unroll
  for (int s = 0; s < CSLICES; ++s) {
    p = fmin_raw(p, posPart[(size_t)s * NROWS + row]);
    n = fmax_raw(n, negPart[(size_t)s * NROWS + row]);
  }
  float l = n - p + MARGIN;
  l = l > 0.f ? l : 0.f;

#pragma unroll
  for (int m = 16; m >= 1; m >>= 1) l += __shfl_xor(l, m, 32);
  __shared__ float ws[8];
  if ((tid & 31) == 0) ws[tid >> 5] = l;
  __syncthreads();
  if (tid < 8) {
    float t = ws[tid];
#pragma unroll
    for (int m = 4; m >= 1; m >>= 1) t += __shfl_xor(t, m, 32);
    if (tid == 0) partials[blockIdx.x] = t;
  }
}

// ---------------------------------------------------------------------------
// Kernel 4: sum 32 partials -> mean loss.
// ---------------------------------------------------------------------------
__global__ void final_reduce(const float* __restrict__ partials,
                             float* __restrict__ out) {
  float s = partials[threadIdx.x];
#pragma unroll
  for (int m = 16; m >= 1; m >>= 1) s += __shfl_xor(s, m, 32);
  if (threadIdx.x == 0) out[0] = s / (float)NROWS;
}

// ---------------------------------------------------------------------------
extern "C" void kernel_launch(void* const* d_in, const int* in_sizes, int n_in,
                              void* d_out, int out_size, void* d_ws, size_t ws_size,
                              hipStream_t stream) {
  const float* features = (const float*)d_in[0];
  const int*   labels   = (const int*)d_in[1];
  float*       out      = (float*)d_out;

  // d_ws layout:
  //   [0, 4 MB)            normalized f16 features
  //   [+0,   +256 KB)      posPart  [CSLICES][NROWS]
  //   [+256, +512 KB)      negPart  [CSLICES][NROWS]
  //   [+512 KB, +128 B)    partials [32]
  char* ws = (char*)d_ws;
  _Float16* fh      = (_Float16*)ws;
  float* posPart    = (float*)(ws + (size_t)NROWS * DIM * 2);
  float* negPart    = posPart + (size_t)CSLICES * NROWS;
  float* partials   = negPart + (size_t)CSLICES * NROWS;

  normalize_f16<<<NROWS / 8, 256, 0, stream>>>(features, fh);
  triplet_tile<<<dim3(NROWS / ROWS_PER_BLOCK, CSLICES), 256, 0, stream>>>(
      fh, labels, posPart, negPart);
  row_loss<<<NROWS / 256, 256, 0, stream>>>(posPart, negPart, partials);
  final_reduce<<<1, 32, 0, stream>>>(partials, out);
}
